// GraphVAE_9740985827608
// MI455X (gfx1250) — matrix operand order repaired and checked by reading, not compiled, not run on hardware
//
#include <hip/hip_runtime.h>
#include <hip/hip_bf16.h>
#include <math.h>

typedef __attribute__((ext_vector_type(16))) __bf16 v16bf;
typedef __attribute__((ext_vector_type(8)))  __bf16 v8bf;
typedef __attribute__((ext_vector_type(8)))  float  v8f;

#define D_IN 512
#define H1D  256
#define LATD 128

// ---------------- elementwise helpers ----------------

__global__ void k_f32_to_bf16(const float* __restrict__ in, __bf16* __restrict__ out, int n) {
    int i = blockIdx.x * blockDim.x + threadIdx.x;
    if (i < n) out[i] = (__bf16)in[i];
}

// W [K x Nn] f32 row-major  ->  Wt [Nn x K] bf16 row-major
__global__ void k_transpose_bf16(const float* __restrict__ W, __bf16* __restrict__ Wt, int K, int Nn) {
    int i = blockIdx.x * blockDim.x + threadIdx.x;
    if (i >= K * Nn) return;
    int k = i / Nn, n = i % Nn;
    Wt[(size_t)n * K + k] = (__bf16)W[i];
}

__global__ void k_fill_f32(float* __restrict__ p, float v, int n) {
    int i = blockIdx.x * blockDim.x + threadIdx.x;
    if (i < n) p[i] = v;
}

__global__ void k_deg_count(const int* __restrict__ col, float* __restrict__ deg, int E) {
    int i = blockIdx.x * blockDim.x + threadIdx.x;
    if (i < E) unsafeAtomicAdd(&deg[col[i]], 1.0f);
}

__global__ void k_rsqrt_inplace(float* __restrict__ p, int n) {
    int i = blockIdx.x * blockDim.x + threadIdx.x;
    if (i < n) p[i] = rsqrtf(p[i]);
}

// ---------------- WMMA bf16 GEMM: C[M x Nn] = A[M x K] * Bt[Nn x K]^T ----------------
// Block: 128 threads = 4 waves. Wave w computes rows [by*64 + 16w, +16) x cols [bx*64, +64)
// (4 N-subtiles of 16, reusing the A fragment).  All dims multiples of 64 (M,N) / 32 (K).

template<bool HAS_BIAS, bool SIGMOID, bool BF16_OUT>
__global__ void k_gemm_wmma(const __bf16* __restrict__ A, const __bf16* __restrict__ Bt,
                            const float* __restrict__ bias, float* __restrict__ C,
                            __bf16* __restrict__ Cbf, int M, int Nn, int K) {
    const int lane = threadIdx.x & 31;
    const int wave = threadIdx.x >> 5;
    const int half = lane >> 4;     // 0 or 1
    const int l16  = lane & 15;
    const int tileM = blockIdx.y * 64 + wave * 16;
    const int tileN = blockIdx.x * 64;

    v8f acc[4] = {};

    // A: lane holds row m = tileM + l16; 16 contiguous K starting at half*16
    const __bf16* ap = A + (size_t)(tileM + l16) * K + half * 16;
    // B: lane holds col n; 16 contiguous K starting at half*16
    const __bf16* bp0 = Bt + (size_t)(tileN + 0 * 16 + l16) * K + half * 16;
    const __bf16* bp1 = Bt + (size_t)(tileN + 1 * 16 + l16) * K + half * 16;
    const __bf16* bp2 = Bt + (size_t)(tileN + 2 * 16 + l16) * K + half * 16;
    const __bf16* bp3 = Bt + (size_t)(tileN + 3 * 16 + l16) * K + half * 16;

    for (int k0 = 0; k0 < K; k0 += 32) {
        v16bf af = *(const v16bf*)(ap + k0);

        v16bf bf0 = *(const v16bf*)(bp0 + k0);
        v16bf bf1 = *(const v16bf*)(bp1 + k0);
        v16bf bf2 = *(const v16bf*)(bp2 + k0);
        v16bf bf3 = *(const v16bf*)(bp3 + k0);

        acc[0] = __builtin_amdgcn_wmma_f32_16x16x32_bf16(false, af, false, bf0, (short)0, acc[0], false, false);
        acc[1] = __builtin_amdgcn_wmma_f32_16x16x32_bf16(false, af, false, bf1, (short)0, acc[1], false, false);
        acc[2] = __builtin_amdgcn_wmma_f32_16x16x32_bf16(false, af, false, bf2, (short)0, acc[2], false, false);
        acc[3] = __builtin_amdgcn_wmma_f32_16x16x32_bf16(false, af, false, bf3, (short)0, acc[3], false, false);
    }

    // C/D layout: VGPR r -> row = tileM + half*8 + r ; lane l16 -> col
#pragma unroll
    for (int t = 0; t < 4; ++t) {
        int col = tileN + t * 16 + l16;
        float bv = HAS_BIAS ? bias[col] : 0.0f;
#pragma unroll
        for (int r = 0; r < 8; ++r) {
            int row = tileM + half * 8 + r;
            float v = acc[t][r] + bv;
            if (SIGMOID) v = 1.0f / (1.0f + expf(-v));
            C[(size_t)row * Nn + col] = v;
            if (BF16_OUT) Cbf[(size_t)row * Nn + col] = (__bf16)v;
        }
    }
}

// ---------------- edge scatter: Agg[c] += dis[r]*dis[c] * H[r] ----------------

__global__ void k_aggregate(const int* __restrict__ rows, const int* __restrict__ cols,
                            const float* __restrict__ dis, const float* __restrict__ H,
                            float* __restrict__ Agg, int E, int F) {
    unsigned long long idx = (unsigned long long)blockIdx.x * blockDim.x + threadIdx.x;
    if (idx >= (unsigned long long)E * (unsigned long long)F) return;
    int e = (int)(idx / (unsigned)F);
    int f = (int)(idx % (unsigned)F);
    int r = rows[e], c = cols[e];
    float nv = dis[r] * dis[c];
    unsafeAtomicAdd(&Agg[(size_t)c * F + f], nv * H[(size_t)r * F + f]);
}

// ---------------- self-loops (x2) + bias (+relu) epilogue, emits bf16 ----------------

template<bool RELU>
__global__ void k_finish(const float* __restrict__ Agg, const float* __restrict__ H,
                         const float* __restrict__ dis, const float* __restrict__ bias,
                         __bf16* __restrict__ Zbf, int Nn, int F) {
    int i = blockIdx.x * blockDim.x + threadIdx.x;
    if (i >= Nn * F) return;
    int n = i / F, f = i % F;
    float d = dis[n];
    float v = Agg[i] + 2.0f * d * d * H[i] + bias[f];
    if (RELU) v = fmaxf(v, 0.0f);
    Zbf[i] = (__bf16)v;
}

// ---------------- launcher ----------------

extern "C" void kernel_launch(void* const* d_in, const int* in_sizes, int n_in,
                              void* d_out, int out_size, void* d_ws, size_t ws_size,
                              hipStream_t stream) {
    const float* x   = (const float*)d_in[0];
    const int*   ei  = (const int*)d_in[1];
    const float* W1  = (const float*)d_in[2];
    const float* b1  = (const float*)d_in[3];
    const float* W2  = (const float*)d_in[4];
    const float* b2  = (const float*)d_in[5];
    const float* Wmu = (const float*)d_in[6];
    const float* bmu = (const float*)d_in[7];
    const float* Wlv = (const float*)d_in[8];
    const float* blv = (const float*)d_in[9];

    const int N = in_sizes[0] / D_IN;   // 8192
    const int E = in_sizes[1] / 2;      // 262144
    const int* e_row = ei;
    const int* e_col = ei + E;

    // workspace carve-out (256B aligned slots)
    size_t off = 0;
    auto alloc = [&](size_t bytes) -> void* {
        void* p = (char*)d_ws + off;
        off += (bytes + 255) & ~(size_t)255;
        return p;
    };
    __bf16* xbf   = (__bf16*)alloc((size_t)N * D_IN * 2);
    __bf16* W1t   = (__bf16*)alloc((size_t)H1D * D_IN * 2);
    __bf16* W2t   = (__bf16*)alloc((size_t)LATD * H1D * 2);
    __bf16* Wmut  = (__bf16*)alloc((size_t)LATD * LATD * 2);
    __bf16* Wlvt  = (__bf16*)alloc((size_t)LATD * LATD * 2);
    float*  dis   = (float*) alloc((size_t)N * 4);
    float*  h1    = (float*) alloc((size_t)N * H1D * 4);
    float*  agg1  = (float*) alloc((size_t)N * H1D * 4);
    __bf16* z1bf  = (__bf16*)alloc((size_t)N * H1D * 2);
    float*  h2    = (float*) alloc((size_t)N * LATD * 4);
    float*  agg2  = (float*) alloc((size_t)N * LATD * 4);
    __bf16* zbf   = (__bf16*)alloc((size_t)N * LATD * 2);
    __bf16* mubf  = (__bf16*)alloc((size_t)N * LATD * 2);

    float* out_adj = (float*)d_out;                       // [N x N]
    float* out_mu  = out_adj + (size_t)N * N;             // [N x LAT]
    float* out_lv  = out_mu  + (size_t)N * LATD;          // [N x LAT]

    const int T = 256;
    auto nb = [&](long long n) { return (unsigned)((n + T - 1) / T); };

    // 1. precision conversions / weight transposes
    hipLaunchKernelGGL(k_f32_to_bf16, dim3(nb((long long)N * D_IN)), dim3(T), 0, stream, x, xbf, N * D_IN);
    hipLaunchKernelGGL(k_transpose_bf16, dim3(nb((long long)D_IN * H1D)), dim3(T), 0, stream, W1, W1t, D_IN, H1D);
    hipLaunchKernelGGL(k_transpose_bf16, dim3(nb((long long)H1D * LATD)), dim3(T), 0, stream, W2, W2t, H1D, LATD);
    hipLaunchKernelGGL(k_transpose_bf16, dim3(nb((long long)LATD * LATD)), dim3(T), 0, stream, Wmu, Wmut, LATD, LATD);
    hipLaunchKernelGGL(k_transpose_bf16, dim3(nb((long long)LATD * LATD)), dim3(T), 0, stream, Wlv, Wlvt, LATD, LATD);

    // 2. degrees: deg[v] = indeg(v) + 2 (self-loops added twice) ; dis = deg^-0.5
    hipLaunchKernelGGL(k_fill_f32, dim3(nb(N)), dim3(T), 0, stream, dis, 2.0f, N);
    hipLaunchKernelGGL(k_deg_count, dim3(nb(E)), dim3(T), 0, stream, e_col, dis, E);
    hipLaunchKernelGGL(k_rsqrt_inplace, dim3(nb(N)), dim3(T), 0, stream, dis, N);

    // 3. layer 1: h1 = x @ W1 (no bias yet) ; scatter ; z1 = relu(agg + 2 dis^2 h1 + b1)
    hipLaunchKernelGGL((k_gemm_wmma<false, false, false>), dim3(H1D / 64, N / 64), dim3(128), 0, stream,
                       xbf, W1t, (const float*)nullptr, h1, (__bf16*)nullptr, N, H1D, D_IN);
    hipLaunchKernelGGL(k_fill_f32, dim3(nb((long long)N * H1D)), dim3(T), 0, stream, agg1, 0.0f, N * H1D);
    hipLaunchKernelGGL(k_aggregate, dim3(nb((long long)E * H1D)), dim3(T), 0, stream,
                       e_row, e_col, dis, h1, agg1, E, H1D);
    hipLaunchKernelGGL((k_finish<true>), dim3(nb((long long)N * H1D)), dim3(T), 0, stream,
                       agg1, h1, dis, b1, z1bf, N, H1D);

    // 4. layer 2: h2 = z1 @ W2 ; scatter ; z = agg + 2 dis^2 h2 + b2
    hipLaunchKernelGGL((k_gemm_wmma<false, false, false>), dim3(LATD / 64, N / 64), dim3(128), 0, stream,
                       z1bf, W2t, (const float*)nullptr, h2, (__bf16*)nullptr, N, LATD, H1D);
    hipLaunchKernelGGL(k_fill_f32, dim3(nb((long long)N * LATD)), dim3(T), 0, stream, agg2, 0.0f, N * LATD);
    hipLaunchKernelGGL(k_aggregate, dim3(nb((long long)E * LATD)), dim3(T), 0, stream,
                       e_row, e_col, dis, h2, agg2, E, LATD);
    hipLaunchKernelGGL((k_finish<false>), dim3(nb((long long)N * LATD)), dim3(T), 0, stream,
                       agg2, h2, dis, b2, zbf, N, LATD);

    // 5. mu = z @ Wmu + bmu (f32 out + bf16 copy) ; logvar = z @ Wlv + blv
    hipLaunchKernelGGL((k_gemm_wmma<true, false, true>), dim3(LATD / 64, N / 64), dim3(128), 0, stream,
                       zbf, Wmut, bmu, out_mu, mubf, N, LATD, LATD);
    hipLaunchKernelGGL((k_gemm_wmma<true, false, false>), dim3(LATD / 64, N / 64), dim3(128), 0, stream,
                       zbf, Wlvt, blv, out_lv, (__bf16*)nullptr, N, LATD, LATD);

    // 6. adj = sigmoid(mu @ mu^T) — dominant GEMM, mu_bf16 (2MB) is L2-resident
    hipLaunchKernelGGL((k_gemm_wmma<false, true, false>), dim3(N / 64, N / 64), dim3(128), 0, stream,
                       mubf, mubf, (const float*)nullptr, out_adj, (__bf16*)nullptr, N, N, LATD);
}
